// TimeSeriesForecaster_10187662426604
// MI455X (gfx1250) — compile-verified, matrix operand order checked by reading, not run on hardware
//
#include <hip/hip_runtime.h>
#include <hip/hip_bf16.h>
#include <math.h>

typedef __bf16 bf16_t;
typedef __attribute__((ext_vector_type(16))) __bf16 v16bf;
typedef __attribute__((ext_vector_type(8)))  float  v8f;

// ---- problem constants ----
constexpr int B_      = 8;
constexpr int L_      = 512;
constexpr int ENC_IN  = 21;
constexpr int C_OUT   = 21;
constexpr int D_MODEL = 512;
constexpr int D_STATE = 16;
constexpr int D_CONV  = 4;
constexpr int D_INNER = 1024;
constexpr int DT_RANK = 32;
constexpr int PRED    = 96;
constexpr int M_ROWS  = B_ * L_;          // 4096

// =====================================================================
// 1) per-(b,c) mean / stdev over L
// =====================================================================
__global__ void stats_kernel(const float* __restrict__ x_enc,
                             float* __restrict__ mean, float* __restrict__ stdev) {
    int bc = blockIdx.x;                    // 0..167
    int b = bc / ENC_IN, c = bc % ENC_IN;
    __shared__ float ssum[256], ssq[256];
    float s = 0.f, q = 0.f;
    for (int l = threadIdx.x; l < L_; l += 256) {
        float v = x_enc[((size_t)b * L_ + l) * ENC_IN + c];
        s += v; q += v * v;
    }
    ssum[threadIdx.x] = s; ssq[threadIdx.x] = q;
    __syncthreads();
    for (int off = 128; off > 0; off >>= 1) {
        if (threadIdx.x < off) {
            ssum[threadIdx.x] += ssum[threadIdx.x + off];
            ssq[threadIdx.x]  += ssq[threadIdx.x + off];
        }
        __syncthreads();
    }
    if (threadIdx.x == 0) {
        float mu  = ssum[0] / (float)L_;
        float var = ssq[0] / (float)L_ - mu * mu;
        mean[bc]  = mu;
        stdev[bc] = sqrtf(var + 1e-5f);
    }
}

// =====================================================================
// 2) emb = ((x-mu)/sigma) @ W_emb + b_emb  -> bf16  (K=21, scalar path)
// =====================================================================
__global__ void embed_kernel(const float* __restrict__ x_enc,
                             const float* __restrict__ W_emb,
                             const float* __restrict__ b_emb,
                             const float* __restrict__ mean,
                             const float* __restrict__ stdev,
                             bf16_t* __restrict__ emb_bf) {
    size_t idx = (size_t)blockIdx.x * 256 + threadIdx.x;  // B*L*D_MODEL
    size_t row = idx >> 9;                                 // /512
    int dcol = (int)(idx & 511);
    int b = (int)(row / L_);
    float acc = b_emb[dcol];
    #pragma unroll 7
    for (int c = 0; c < ENC_IN; ++c) {
        float xn = (x_enc[row * ENC_IN + c] - mean[b * ENC_IN + c]) / stdev[b * ENC_IN + c];
        acc += xn * W_emb[c * D_MODEL + dcol];
    }
    emb_bf[idx] = (bf16_t)acc;
}

// =====================================================================
// 3) weight convert f32 (KxN row-major) -> bf16 TRANSPOSED (NxK row-major)
//    so the WMMA B operand becomes contiguous-per-lane like A.
// =====================================================================
__global__ void cvt_bf16_t_kernel(const float* __restrict__ in, bf16_t* __restrict__ out,
                                  int K, int N) {
    int i = blockIdx.x * 256 + threadIdx.x;       // over K*N
    if (i >= K * N) return;
    int k = i / N, n = i % N;
    out[(size_t)n * K + k] = (bf16_t)in[i];
}

// slice first 32 of every 64-wide dbc row -> bf16 A-operand for dt GEMM
__global__ void slice_dt_kernel(const float* __restrict__ dbc, bf16_t* __restrict__ dtr) {
    int i = blockIdx.x * 256 + threadIdx.x;      // M_ROWS*32
    int row = i >> 5, c = i & 31;
    dtr[i] = (bf16_t)dbc[(size_t)row * 64 + c];
}

// =====================================================================
// 4) bf16 WMMA GEMM:  C(MxN,f32) = A(MxK) * Bt(NxK)^T, both row-major.
//    8 waves/block; each wave computes one 16x64 strip (4 N-tiles),
//    reusing the A fragment across 4 v_wmma_f32_16x16x32_bf16 per k-step.
// =====================================================================
__global__ void wmma_gemm_bf16_nt(const bf16_t* __restrict__ A,
                                  const bf16_t* __restrict__ Bt,
                                  float* __restrict__ C,
                                  int M, int N, int K) {
    const int wave  = threadIdx.x >> 5;
    const int lane  = threadIdx.x & 31;
    const int col0  = blockIdx.x * 64;            // 4 N-tiles per wave
    const int tileM = blockIdx.y * 8 + wave;      // M/16 tiles (8 waves per block)
    if (tileM * 16 >= M) return;                  // wave-uniform; EXEC stays all-1s
    const int row0 = tileM * 16;
    const int mn = lane & 15;                     // A: M row / B: N col
    const int hi = lane >> 4;                     // half-wave K shift

    const bf16_t* aRow = A + (size_t)(row0 + mn) * K + hi * 8;
    const bf16_t* bRow0 = Bt + (size_t)(col0 + mn) * K + hi * 8;

    v8f acc0 = {}, acc1 = {}, acc2 = {}, acc3 = {};
    for (int k0 = 0; k0 < K; k0 += 32) {
        if (k0 + 32 < K) {                         // prefetch next K slice
            __builtin_prefetch(aRow + k0 + 32, 0, 0);
            __builtin_prefetch(bRow0 + k0 + 32, 0, 0);
        }
        v16bf a;
        #pragma unroll
        for (int e = 0; e < 8; ++e) { a[e] = aRow[k0 + e]; a[8 + e] = aRow[k0 + 16 + e]; }

        #pragma unroll
        for (int nt = 0; nt < 4; ++nt) {
            const bf16_t* bp = bRow0 + (size_t)nt * 16 * K + k0;
            v16bf b;
            #pragma unroll
            for (int e = 0; e < 8; ++e) { b[e] = bp[e]; b[8 + e] = bp[16 + e]; }
            v8f* accp = (nt == 0) ? &acc0 : (nt == 1) ? &acc1 : (nt == 2) ? &acc2 : &acc3;
            *accp = __builtin_amdgcn_wmma_f32_16x16x32_bf16(
                false, a, false, b, (short)0, *accp, false, false);
        }
    }
    #pragma unroll
    for (int nt = 0; nt < 4; ++nt) {              // C/D layout: VGPR r -> row r + 8*hi
        v8f acc = (nt == 0) ? acc0 : (nt == 1) ? acc1 : (nt == 2) ? acc2 : acc3;
        #pragma unroll
        for (int r = 0; r < 8; ++r) {
            C[(size_t)(row0 + r + hi * 8) * N + col0 + nt * 16 + mn] = acc[r];
        }
    }
}

// =====================================================================
// 5) depthwise causal conv (k=4) + SiLU on xs half of xz
// =====================================================================
__global__ void conv_silu_kernel(const float* __restrict__ xz,
                                 const float* __restrict__ conv_w,
                                 const float* __restrict__ conv_b,
                                 float* __restrict__ xs_f,
                                 bf16_t* __restrict__ xs_bf) {
    size_t i = (size_t)blockIdx.x * 256 + threadIdx.x;  // M_ROWS * D_INNER
    size_t row = i >> 10;                               // /1024
    int d = (int)(i & 1023);
    int l = (int)(row & (L_ - 1));
    size_t base = row - l;                              // b*L
    float acc = conv_b[d];
    #pragma unroll
    for (int k = 0; k < D_CONV; ++k) {
        int li = l - (D_CONV - 1) + k;
        if (li >= 0) acc += xz[(base + li) * (2 * D_INNER) + d] * conv_w[d * D_CONV + k];
    }
    float v = acc / (1.f + __expf(-acc));               // SiLU
    xs_f[i]  = v;
    xs_bf[i] = (bf16_t)v;
}

// =====================================================================
// 6) selective scan: one thread per (b,d); h[16] + A-row in registers,
//    B/C vectors broadcast via LDS each timestep
// =====================================================================
__global__ void scan_kernel(const float* __restrict__ dbc,    // (M,64): [32:48]=B, [48:64]=C
                            const float* __restrict__ dtg,    // raw dt GEMM (M, D_INNER)
                            const float* __restrict__ xs_f,   // (M, D_INNER)
                            const float* __restrict__ xz,     // (M, 2*D_INNER) for z half
                            const float* __restrict__ b_dt,
                            const float* __restrict__ A_log,
                            const float* __restrict__ Dvec,
                            bf16_t* __restrict__ yz_bf) {     // (M, D_INNER) bf16
    __shared__ float sB[D_STATE], sC[D_STATE];
    const int b = blockIdx.y;
    const int d = blockIdx.x * blockDim.x + threadIdx.x;
    float As[D_STATE], h[D_STATE];
    #pragma unroll
    for (int s = 0; s < D_STATE; ++s) {
        As[s] = -__expf(A_log[d * D_STATE + s]);
        h[s]  = 0.f;
    }
    const float bdt = b_dt[d], Dd = Dvec[d];
    for (int l = 0; l < L_; ++l) {
        size_t rl = (size_t)b * L_ + l;
        if (threadIdx.x < 2 * D_STATE) {
            float v = dbc[rl * 64 + DT_RANK + threadIdx.x];
            if (threadIdx.x < D_STATE) sB[threadIdx.x] = v;
            else                       sC[threadIdx.x - D_STATE] = v;
        }
        __syncthreads();
        float draw = dtg[rl * D_INNER + d] + bdt;
        float dt = (draw > 20.f) ? draw : log1pf(__expf(draw));   // softplus
        float x = xs_f[rl * D_INNER + d];
        float dtx = dt * x;
        float y = 0.f;
        #pragma unroll
        for (int s = 0; s < D_STATE; ++s) {
            float dA = __expf(dt * As[s]);
            h[s] = h[s] * dA + dtx * sB[s];
            y += h[s] * sC[s];
        }
        y += x * Dd;
        float z = xz[rl * (2 * D_INNER) + D_INNER + d];
        float o = y * (z / (1.f + __expf(-z)));                   // y * silu(z)
        yz_bf[rl * D_INNER + d] = (bf16_t)o;
        __syncthreads();
    }
}

// =====================================================================
// 7) head: out[b,t,c] = (proj[b,416+t,:] . W_head[:,c] + b_head[c])*std + mu
// =====================================================================
__global__ void head_kernel(const float* __restrict__ yproj,
                            const float* __restrict__ W_head,
                            const float* __restrict__ b_head,
                            const float* __restrict__ mean,
                            const float* __restrict__ stdev,
                            float* __restrict__ out) {
    int i = blockIdx.x * 256 + threadIdx.x;       // B*PRED*C_OUT = 16128
    if (i >= B_ * PRED * C_OUT) return;
    int c = i % C_OUT;
    int t = (i / C_OUT) % PRED;
    int b = i / (C_OUT * PRED);
    const float* row = yproj + ((size_t)b * L_ + (L_ - PRED) + t) * D_MODEL;
    float acc = b_head[c];
    for (int m = 0; m < D_MODEL; ++m) acc += row[m] * W_head[m * C_OUT + c];
    out[i] = acc * stdev[b * C_OUT + c] + mean[b * C_OUT + c];
}

// =====================================================================
// launch
// =====================================================================
static inline size_t align256(size_t x) { return (x + 255) & ~(size_t)255; }

extern "C" void kernel_launch(void* const* d_in, const int* in_sizes, int n_in,
                              void* d_out, int out_size, void* d_ws, size_t ws_size,
                              hipStream_t stream) {
    const float* x_enc  = (const float*)d_in[0];
    const float* W_emb  = (const float*)d_in[4];
    const float* b_emb  = (const float*)d_in[5];
    const float* W_in   = (const float*)d_in[6];
    const float* conv_w = (const float*)d_in[7];
    const float* conv_b = (const float*)d_in[8];
    const float* W_xp   = (const float*)d_in[9];
    const float* W_dt   = (const float*)d_in[10];
    const float* b_dt   = (const float*)d_in[11];
    const float* A_log  = (const float*)d_in[12];
    const float* Dvec   = (const float*)d_in[13];
    const float* W_op   = (const float*)d_in[14];
    const float* W_head = (const float*)d_in[15];
    const float* b_head = (const float*)d_in[16];
    float* out = (float*)d_out;

    // ---- workspace carve-up ----
    char* ws = (char*)d_ws;
    size_t off = 0;
    auto take = [&](size_t bytes) { char* p = ws + off; off = align256(off + bytes); return p; };
    float*  mean_p   = (float*) take(B_ * ENC_IN * 4);
    float*  stdev_p  = (float*) take(B_ * ENC_IN * 4);
    bf16_t* emb_bf   = (bf16_t*)take((size_t)M_ROWS * D_MODEL * 2);
    bf16_t* W_in_bt  = (bf16_t*)take((size_t)D_MODEL * 2 * D_INNER * 2);   // (2048 x 512)
    float*  xz_f     = (float*) take((size_t)M_ROWS * 2 * D_INNER * 4);
    float*  xs_f     = (float*) take((size_t)M_ROWS * D_INNER * 4);
    bf16_t* xs_bf    = (bf16_t*)take((size_t)M_ROWS * D_INNER * 2);
    bf16_t* W_xp_bt  = (bf16_t*)take((size_t)D_INNER * 64 * 2);            // (64 x 1024)
    float*  dbc_f    = (float*) take((size_t)M_ROWS * 64 * 4);
    bf16_t* dtr_bf   = (bf16_t*)take((size_t)M_ROWS * DT_RANK * 2);
    bf16_t* W_dt_bt  = (bf16_t*)take((size_t)DT_RANK * D_INNER * 2);       // (1024 x 32)
    float*  dtg_f    = (float*) take((size_t)M_ROWS * D_INNER * 4);
    bf16_t* yz_bf    = (bf16_t*)take((size_t)M_ROWS * D_INNER * 2);
    bf16_t* W_op_bt  = (bf16_t*)take((size_t)D_INNER * D_MODEL * 2);       // (512 x 1024)
    float*  yproj_f  = (float*) take((size_t)M_ROWS * D_MODEL * 4);
    (void)ws_size;

    // 1) stats
    stats_kernel<<<B_ * ENC_IN, 256, 0, stream>>>(x_enc, mean_p, stdev_p);

    // 2) embed -> bf16
    embed_kernel<<<(M_ROWS * D_MODEL) / 256, 256, 0, stream>>>(
        x_enc, W_emb, b_emb, mean_p, stdev_p, emb_bf);

    // 3) weight conversions f32 -> bf16, transposed to (N x K)
    cvt_bf16_t_kernel<<<(D_MODEL * 2 * D_INNER) / 256, 256, 0, stream>>>(
        W_in, W_in_bt, D_MODEL, 2 * D_INNER);
    cvt_bf16_t_kernel<<<(D_INNER * 64) / 256, 256, 0, stream>>>(
        W_xp, W_xp_bt, D_INNER, 64);
    cvt_bf16_t_kernel<<<(DT_RANK * D_INNER) / 256, 256, 0, stream>>>(
        W_dt, W_dt_bt, DT_RANK, D_INNER);
    cvt_bf16_t_kernel<<<(D_INNER * D_MODEL) / 256, 256, 0, stream>>>(
        W_op, W_op_bt, D_INNER, D_MODEL);

    // 4) xz = emb @ W_in   (4096 x 512 x 2048)
    wmma_gemm_bf16_nt<<<dim3((2 * D_INNER) / 64, M_ROWS / 128), 256, 0, stream>>>(
        emb_bf, W_in_bt, xz_f, M_ROWS, 2 * D_INNER, D_MODEL);

    // 5) causal depthwise conv + SiLU
    conv_silu_kernel<<<(M_ROWS * D_INNER) / 256, 256, 0, stream>>>(
        xz_f, conv_w, conv_b, xs_f, xs_bf);

    // 6) dbc = xs @ W_xp   (4096 x 1024 x 64)
    wmma_gemm_bf16_nt<<<dim3(64 / 64, M_ROWS / 128), 256, 0, stream>>>(
        xs_bf, W_xp_bt, dbc_f, M_ROWS, 64, D_INNER);

    // 7) dt operand slice, then dt = dbc[:, :32] @ W_dt  (4096 x 32 x 1024)
    slice_dt_kernel<<<(M_ROWS * DT_RANK) / 256, 256, 0, stream>>>(dbc_f, dtr_bf);
    wmma_gemm_bf16_nt<<<dim3(D_INNER / 64, M_ROWS / 128), 256, 0, stream>>>(
        dtr_bf, W_dt_bt, dtg_f, M_ROWS, D_INNER, DT_RANK);

    // 8) selective scan (+ softplus, +x*D, * silu(z)) -> yz bf16
    scan_kernel<<<dim3(D_INNER / 256, B_), 256, 0, stream>>>(
        dbc_f, dtg_f, xs_f, xz_f, b_dt, A_log, Dvec, yz_bf);

    // 9) yproj = yz @ W_op  (4096 x 1024 x 512)
    wmma_gemm_bf16_nt<<<dim3(D_MODEL / 64, M_ROWS / 128), 256, 0, stream>>>(
        yz_bf, W_op_bt, yproj_f, M_ROWS, D_MODEL, D_INNER);

    // 10) head + de-normalization
    head_kernel<<<(B_ * PRED * C_OUT + 255) / 256, 256, 0, stream>>>(
        yproj_f, W_head, b_head, mean_p, stdev_p, out);
}
